// DiffusionRNNAgent_6493990551596
// MI455X (gfx1250) — matrix operand order repaired and checked by reading, not compiled, not run on hardware
//
#include <hip/hip_runtime.h>
#include <hip/hip_bf16.h>
#include <math.h>

#ifndef M_PI
#define M_PI 3.14159265358979323846
#endif

#define BATCH 32768
#define IN_DIM 128
#define RNN_H 256
#define N_ACT 32
#define T_DIMC 16
#define MIDD 256
#define N_T 10

typedef _Float16 h16;
typedef __attribute__((ext_vector_type(16))) _Float16 v16h;
typedef __attribute__((ext_vector_type(8)))  float    v8f;

// ---------------------------------------------------------------------------
// WMMA helpers (CDNA5: D = A(16x32,f16) x B(32x16,f16) + C(16x16,f32))
// ---------------------------------------------------------------------------
__device__ inline v8f wmma32(v16h a, v16h b, v8f c) {
  return __builtin_amdgcn_wmma_f32_16x16x32_f16(
      /*neg_a=*/false, a, /*neg_b=*/false, b,
      /*c_mod=*/(short)0, c, /*reuse_a=*/false, /*reuse_b=*/false);
}

// A fragment (16x32 f16, row-major source). Per ISA layout:
// lanes 0-15:  row = lane,    K = k0+[0..7] and k0+[16..23]
// lanes 16-31: row = lane-16, K = k0+[8..15] and k0+[24..31]
__device__ inline v16h load_a_f16(const h16* A, int lda, int row0, int k0, int lane) {
  int r  = row0 + (lane & 15);
  const h16* p = A + r * lda + k0 + ((lane >> 4) << 3);
  union { uint4 u[2]; v16h v; } t;
  t.u[0] = *(const uint4*)(p);
  t.u[1] = *(const uint4*)(p + 16);
  return t.v;
}

__device__ inline v16h load_a_f32(const float* A, int lda, int row0, int k0, int lane) {
  int r  = row0 + (lane & 15);
  const float* p = A + r * lda + k0 + ((lane >> 4) << 3);
  float4 f0 = ((const float4*)p)[0];
  float4 f1 = ((const float4*)p)[1];
  float4 f2 = ((const float4*)(p + 16))[0];
  float4 f3 = ((const float4*)(p + 16))[1];
  v16h a;
  a[0]=(h16)f0.x; a[1]=(h16)f0.y; a[2]=(h16)f0.z; a[3]=(h16)f0.w;
  a[4]=(h16)f1.x; a[5]=(h16)f1.y; a[6]=(h16)f1.z; a[7]=(h16)f1.w;
  a[8]=(h16)f2.x; a[9]=(h16)f2.y; a[10]=(h16)f2.z; a[11]=(h16)f2.w;
  a[12]=(h16)f3.x; a[13]=(h16)f3.y; a[14]=(h16)f3.z; a[15]=(h16)f3.w;
  return a;
}

// B fragment (32x16 K-major). Same K order as A: lane = n (0-15),
// K = k0+(lane/16)*8+[0..7] and k0+16+(lane/16)*8+[0..7].
// Weight W stored [N][Kpad] row-major (f16) -> B[k][n] = W[n][k].
__device__ inline v16h load_b(const h16* W, int ldw, int n0, int k0, int lane) {
  int n  = n0 + (lane & 15);
  const h16* p = W + n * ldw + k0 + ((lane >> 4) << 3);
  union { uint4 u[2]; v16h v; } t;
  t.u[0] = *(const uint4*)(p);
  t.u[1] = *(const uint4*)(p + 16);
  return t.v;
}

// Concat A source for m1: cols [0,32)=xa f32, [32,48)=te f32 (row-bcast),
// [48,304)=h f16, [304,320)=zero pad. Chunks of 8 never straddle boundaries.
__device__ inline void load8_concat(h16* d, int kc, int row,
                                    const float* xa, const float* te,
                                    const h16* hbuf) {
  if (kc < N_ACT + T_DIMC) {
    const float* p = (kc < N_ACT) ? (xa + row * N_ACT + kc) : (te + (kc - N_ACT));
    float4 f0 = ((const float4*)p)[0];
    float4 f1 = ((const float4*)p)[1];
    d[0]=(h16)f0.x; d[1]=(h16)f0.y; d[2]=(h16)f0.z; d[3]=(h16)f0.w;
    d[4]=(h16)f1.x; d[5]=(h16)f1.y; d[6]=(h16)f1.z; d[7]=(h16)f1.w;
  } else if (kc < N_ACT + T_DIMC + RNN_H) {
    *(uint4*)d = *(const uint4*)(hbuf + row * RNN_H + (kc - N_ACT - T_DIMC));
  } else {
    uint4 z; z.x = 0u; z.y = 0u; z.z = 0u; z.w = 0u;
    *(uint4*)d = z;
  }
}

__device__ inline v16h load_a_concat(int row0, int k0, int lane,
                                     const float* xa, const float* te,
                                     const h16* hbuf) {
  int r  = row0 + (lane & 15);
  int kc = k0 + ((lane >> 4) << 3);
  union { uint4 u[2]; h16 h[16]; v16h v; } t;
  load8_concat(t.h,     kc,      r, xa, te, hbuf);
  load8_concat(t.h + 8, kc + 16, r, xa, te, hbuf);
  return t.v;
}

__device__ inline float mishf(float x) {
  float sp = log1pf(expf(x));
  return x * tanhf(sp);
}

// ---------------------------------------------------------------------------
// Generic 256-wide WMMA GEMM: out16 = act(A[M,Kpad] @ W[256,Kpad]^T + b)
// Block = 64 rows, 8 waves = 4(M) x 2(N); each wave: 16 rows x 128 cols.
// ---------------------------------------------------------------------------
enum { AM_F32 = 0, AM_F16 = 1, AM_CAT = 2 };
enum { ACT_RELU = 0, ACT_MISH = 1 };

template<int AMODE, int ACT>
__global__ void __launch_bounds__(256)
gemm256_k(const float* Af32, const h16* Af16,
          const float* xa, const float* te, const h16* hbuf,
          int Kpad, const h16* W, const float* bias, h16* out16) {
  int lane = threadIdx.x & 31;
  int wave = threadIdx.x >> 5;
  int wm = wave >> 1, wn = wave & 1;
  int row0 = blockIdx.x * 64 + wm * 16;
  int nb = wn * 128;

  v8f acc[8];
#pragma unroll
  for (int t = 0; t < 8; ++t)
#pragma unroll
    for (int r = 0; r < 8; ++r) acc[t][r] = 0.f;

  int nkt = Kpad >> 5;
  for (int kt = 0; kt < nkt; ++kt) {
    int k0 = kt << 5;
    v16h a;
    if constexpr (AMODE == AM_F32)      a = load_a_f32(Af32, Kpad, row0, k0, lane);
    else if constexpr (AMODE == AM_F16) a = load_a_f16(Af16, Kpad, row0, k0, lane);
    else                                a = load_a_concat(row0, k0, lane, xa, te, hbuf);
#pragma unroll
    for (int t = 0; t < 8; ++t) {
      v16h b = load_b(W, Kpad, nb + t * 16, k0, lane);
      acc[t] = wmma32(a, b, acc[t]);
    }
  }

  int m0 = row0 + ((lane >> 4) << 3);
#pragma unroll
  for (int t = 0; t < 8; ++t) {
    int nc = nb + t * 16 + (lane & 15);
    float bv = bias[nc];
#pragma unroll
    for (int r = 0; r < 8; ++r) {
      float v = acc[t][r] + bv;
      if constexpr (ACT == ACT_RELU) v = fmaxf(v, 0.f);
      else                           v = mishf(v);
      out16[(m0 + r) * 256 + nc] = (h16)v;
    }
  }
}

// ---------------------------------------------------------------------------
// Wave-level accumulate helpers for the fused GRU kernel
// ---------------------------------------------------------------------------
__device__ inline void wave_mm_f16(v8f acc[8], const h16* A, int lda, int row0,
                                   const h16* W, int ldw, int n0, int K, int lane) {
  for (int kt = 0; kt < (K >> 5); ++kt) {
    int k0 = kt << 5;
    v16h a = load_a_f16(A, lda, row0, k0, lane);
#pragma unroll
    for (int t = 0; t < 8; ++t) {
      v16h b = load_b(W, ldw, n0 + t * 16, k0, lane);
      acc[t] = wmma32(a, b, acc[t]);
    }
  }
}

__device__ inline void wave_mm_f32(v8f acc[8], const float* A, int lda, int row0,
                                   const h16* W, int ldw, int n0, int K, int lane) {
  for (int kt = 0; kt < (K >> 5); ++kt) {
    int k0 = kt << 5;
    v16h a = load_a_f32(A, lda, row0, k0, lane);
#pragma unroll
    for (int t = 0; t < 8; ++t) {
      v16h b = load_b(W, ldw, n0 + t * 16, k0, lane);
      acc[t] = wmma32(a, b, acc[t]);
    }
  }
}

// ---------------------------------------------------------------------------
// Fused GRUCell: gi = x@Wih^T, gh = h@Whh^T, gates in 128KB dynamic LDS.
// Writes h (f32, to d_out) and f16 copy for the diffusion MLP.
// ---------------------------------------------------------------------------
__global__ void __launch_bounds__(256)
gru_k(const h16* x16, const float* h_in,
      const h16* Wih, const h16* Whh,
      const float* bih, const float* bhh,
      float* h_out, h16* h16out) {
  extern __shared__ h16 lds[];
  h16* g_r  = lds;
  h16* g_z  = lds + 64 * 256;
  h16* g_in = lds + 2 * 64 * 256;
  h16* g_hn = lds + 3 * 64 * 256;

  int lane = threadIdx.x & 31;
  int wave = threadIdx.x >> 5;
  int wm = wave >> 1, wn = wave & 1;
  int row0 = blockIdx.x * 64 + wm * 16;
  int nb = wn * 128;
  int lm0 = wm * 16 + ((lane >> 4) << 3);

  // r and z gates: sigmoid(i_g + h_g)
  for (int g = 0; g < 2; ++g) {
    v8f acc[8];
#pragma unroll
    for (int t = 0; t < 8; ++t)
#pragma unroll
      for (int r = 0; r < 8; ++r) acc[t][r] = 0.f;
    wave_mm_f16(acc, x16,  RNN_H, row0, Wih, RNN_H, g * 256 + nb, RNN_H, lane);
    wave_mm_f32(acc, h_in, RNN_H, row0, Whh, RNN_H, g * 256 + nb, RNN_H, lane);
    h16* dst = (g == 0) ? g_r : g_z;
#pragma unroll
    for (int t = 0; t < 8; ++t) {
      int nc = nb + t * 16 + (lane & 15);
      float b0 = bih[g * 256 + nc] + bhh[g * 256 + nc];
#pragma unroll
      for (int r = 0; r < 8; ++r) {
        float v = acc[t][r] + b0;
        v = 1.f / (1.f + expf(-v));
        dst[(lm0 + r) * 256 + nc] = (h16)v;
      }
    }
  }
  // i_n (kept separate from h_n: n = tanh(i_n + r*h_n))
  {
    v8f acc[8];
#pragma unroll
    for (int t = 0; t < 8; ++t)
#pragma unroll
      for (int r = 0; r < 8; ++r) acc[t][r] = 0.f;
    wave_mm_f16(acc, x16, RNN_H, row0, Wih, RNN_H, 512 + nb, RNN_H, lane);
#pragma unroll
    for (int t = 0; t < 8; ++t) {
      int nc = nb + t * 16 + (lane & 15);
      float b0 = bih[512 + nc];
#pragma unroll
      for (int r = 0; r < 8; ++r)
        g_in[(lm0 + r) * 256 + nc] = (h16)(acc[t][r] + b0);
    }
  }
  // h_n
  {
    v8f acc[8];
#pragma unroll
    for (int t = 0; t < 8; ++t)
#pragma unroll
      for (int r = 0; r < 8; ++r) acc[t][r] = 0.f;
    wave_mm_f32(acc, h_in, RNN_H, row0, Whh, RNN_H, 512 + nb, RNN_H, lane);
#pragma unroll
    for (int t = 0; t < 8; ++t) {
      int nc = nb + t * 16 + (lane & 15);
      float b0 = bhh[512 + nc];
#pragma unroll
      for (int r = 0; r < 8; ++r)
        g_hn[(lm0 + r) * 256 + nc] = (h16)(acc[t][r] + b0);
    }
  }
  __syncthreads();
  for (int e = threadIdx.x; e < 64 * 256; e += 256) {
    int ml = e >> 8, c = e & 255;
    int grow = blockIdx.x * 64 + ml;
    float rv = (float)g_r[e], zv = (float)g_z[e];
    float nv = tanhf((float)g_in[e] + rv * (float)g_hn[e]);
    float hp = h_in[grow * 256 + c];
    float hv = (1.f - zv) * nv + zv * hp;
    h_out[grow * 256 + c]  = hv;
    h16out[grow * 256 + c] = (h16)hv;
  }
}

// ---------------------------------------------------------------------------
// fin head (256->32) with fused diffusion posterior update.
// Block = 64 rows, 8 waves = 4(M) x 2(N); each wave 16 rows x 16 cols.
// ---------------------------------------------------------------------------
__global__ void __launch_bounds__(256)
fin_k(const h16* A, const h16* W, const float* bias,
      const float* xa_in, const float* noise,
      float* xa_out, float* q_out,
      float sr, float srm1, float c1, float c2, float sigma, int is_final) {
  int lane = threadIdx.x & 31;
  int wave = threadIdx.x >> 5;
  int wm = wave >> 1, wn = wave & 1;
  int row0 = blockIdx.x * 64 + wm * 16;
  int n0 = wn * 16;

  v8f acc;
#pragma unroll
  for (int r = 0; r < 8; ++r) acc[r] = 0.f;
  for (int kt = 0; kt < 8; ++kt) {
    int k0 = kt << 5;
    v16h a = load_a_f16(A, MIDD, row0, k0, lane);
    v16h b = load_b(W, MIDD, n0, k0, lane);
    acc = wmma32(a, b, acc);
  }
  int nc = n0 + (lane & 15);
  int m0 = row0 + ((lane >> 4) << 3);
  float bv = bias[nc];
#pragma unroll
  for (int r = 0; r < 8; ++r) {
    int m = m0 + r;
    float eps = acc[r] + bv;
    float x = xa_in[m * N_ACT + nc];
    float xr = fminf(1.f, fmaxf(-1.f, sr * x - srm1 * eps));
    float mean = c1 * xr + c2 * x;
    if (is_final) q_out[m * N_ACT + nc] = mean;
    else          xa_out[m * N_ACT + nc] = mean + sigma * noise[m * N_ACT + nc];
  }
}

// ---------------------------------------------------------------------------
// Small kernels: weight f32->f16 (+K pad), time-embedding MLP, output tail.
// ---------------------------------------------------------------------------
__global__ void cvt_pad_k(const float* src, h16* dst, int N, int K, int Kpad) {
  int t = blockIdx.x * blockDim.x + threadIdx.x;
  if (t >= N * Kpad) return;
  int n = t / Kpad, k = t % Kpad;
  dst[t] = (k < K) ? (h16)src[n * K + k] : (h16)0.f;
}

__global__ void time_mlp_k(const float* tm1w, const float* tm1b,
                           const float* tm2w, const float* tm2b,
                           float* te_all) {
  __shared__ float emb[16];
  __shared__ float t1[32];
  int t = threadIdx.x;  // 32 threads
  for (int i = 0; i < N_T; ++i) {
    if (t < 8) {
      float f = expf(-logf(10000.f) * (float)t / 7.f);
      float ang = (float)i * f;
      emb[t]     = sinf(ang);
      emb[t + 8] = cosf(ang);
    }
    __syncthreads();
    if (t < 32) {
      float s = tm1b[t];
      for (int j = 0; j < 16; ++j) s += emb[j] * tm1w[t * 16 + j];
      t1[t] = mishf(s);
    }
    __syncthreads();
    if (t < 16) {
      float s = tm2b[t];
      for (int j = 0; j < 32; ++j) s += t1[j] * tm2w[t * 32 + j];
      te_all[i * 16 + t] = s;
    }
    __syncthreads();
  }
}

__global__ void tail_k(const float* noise9, float* qlog, float* mask,
                       float* noise_out, float qlogv) {
  int t = blockIdx.x * blockDim.x + threadIdx.x;
  if (t < BATCH) { qlog[t] = qlogv; mask[t] = 0.f; }
  if (t < BATCH * N_ACT) noise_out[t] = noise9[t];
}

// ---------------------------------------------------------------------------
extern "C" void kernel_launch(void* const* d_in, const int* in_sizes, int n_in,
                              void* d_out, int out_size, void* d_ws, size_t ws_size,
                              hipStream_t stream) {
  (void)in_sizes; (void)n_in; (void)out_size; (void)ws_size;

  const float* inputs     = (const float*)d_in[0];
  const float* hidden     = (const float*)d_in[1];
  const float* fc1_w      = (const float*)d_in[2];
  const float* fc1_b      = (const float*)d_in[3];
  const float* gwih       = (const float*)d_in[4];
  const float* gwhh       = (const float*)d_in[5];
  const float* gbih       = (const float*)d_in[6];
  const float* gbhh       = (const float*)d_in[7];
  const float* tm1w       = (const float*)d_in[8];
  const float* tm1b       = (const float*)d_in[9];
  const float* tm2w       = (const float*)d_in[10];
  const float* tm2b       = (const float*)d_in[11];
  const float* m1w        = (const float*)d_in[12];
  const float* m1b        = (const float*)d_in[13];
  const float* m2w        = (const float*)d_in[14];
  const float* m2b        = (const float*)d_in[15];
  const float* m3w        = (const float*)d_in[16];
  const float* m3b        = (const float*)d_in[17];
  const float* finw       = (const float*)d_in[18];
  const float* finb       = (const float*)d_in[19];
  const float* x_init     = (const float*)d_in[20];
  const float* step_noise = (const float*)d_in[21];

  float* out    = (float*)d_out;
  float* q_out  = out;                              // [B,32]
  float* h_out  = out + (size_t)BATCH * N_ACT;      // [B,256]
  float* qlog   = out + (size_t)BATCH * (N_ACT + RNN_H);
  float* mask   = qlog + BATCH;
  float* noiseo = mask + BATCH;                     // [B,32]

  // workspace layout (all 16B+ aligned)
  char* w = (char*)d_ws;
  h16* fc1w16 = (h16*)w; w += 256 * 128 * 2;
  h16* gwih16 = (h16*)w; w += 768 * 256 * 2;
  h16* gwhh16 = (h16*)w; w += 768 * 256 * 2;
  h16* m1w16  = (h16*)w; w += 256 * 320 * 2;
  h16* m2w16  = (h16*)w; w += 256 * 256 * 2;
  h16* m3w16  = (h16*)w; w += 256 * 256 * 2;
  h16* finw16 = (h16*)w; w += 32 * 256 * 2;
  float* te_all = (float*)w; w += 1024;
  h16* x16  = (h16*)w; w += (size_t)BATCH * 256 * 2;
  h16* hb16 = (h16*)w; w += (size_t)BATCH * 256 * 2;
  h16* a0   = (h16*)w; w += (size_t)BATCH * 256 * 2;
  h16* a1   = (h16*)w; w += (size_t)BATCH * 256 * 2;
  float* xa = (float*)w; w += (size_t)BATCH * N_ACT * 4;

  // weight prep (f32 -> f16, zero K-pad for m1)
  {
    struct { const float* s; h16* d; int N, K, Kp; } cv[7] = {
      {fc1_w, fc1w16, 256, 128, 128},
      {gwih,  gwih16, 768, 256, 256},
      {gwhh,  gwhh16, 768, 256, 256},
      {m1w,   m1w16,  256, 304, 320},
      {m2w,   m2w16,  256, 256, 256},
      {m3w,   m3w16,  256, 256, 256},
      {finw,  finw16,  32, 256, 256},
    };
    for (int i = 0; i < 7; ++i) {
      int n = cv[i].N * cv[i].Kp;
      cvt_pad_k<<<(n + 255) / 256, 256, 0, stream>>>(cv[i].s, cv[i].d,
                                                     cv[i].N, cv[i].K, cv[i].Kp);
    }
  }
  time_mlp_k<<<1, 32, 0, stream>>>(tm1w, tm1b, tm2w, tm2b, te_all);

  dim3 g(BATCH / 64);
  // fc1 + ReLU
  gemm256_k<AM_F32, ACT_RELU><<<g, 256, 0, stream>>>(
      inputs, nullptr, nullptr, nullptr, nullptr, IN_DIM, fc1w16, fc1_b, x16);
  // fused GRUCell (128 KB dynamic LDS for gate staging)
  gru_k<<<g, 256, 4 * 64 * 256 * (int)sizeof(h16), stream>>>(
      x16, hidden, gwih16, gwhh16, gbih, gbhh, h_out, hb16);

  // diffusion constants (host, double precision; mirrors the numpy formula)
  double betas[N_T], alphas[N_T], acum[N_T], acp[N_T];
  {
    double s0 = 0.008, acv[N_T + 1];
    for (int j = 0; j <= N_T; ++j) {
      double xj = (double)j * (double)(N_T + 1) / (double)N_T;  // linspace(0,11,11)
      double c = cos(((xj / (double)(N_T + 1)) + s0) / (1.0 + s0) * M_PI * 0.5);
      acv[j] = c * c;
    }
    for (int t = 0; t < N_T; ++t) {
      double b = 1.0 - (acv[t + 1] / acv[0]) / (acv[t] / acv[0]);
      if (b < 0.0) b = 0.0;
      if (b > 0.999) b = 0.999;
      betas[t] = b; alphas[t] = 1.0 - b;
    }
    double c = 1.0;
    for (int t = 0; t < N_T; ++t) { c *= alphas[t]; acum[t] = c; }
    for (int t = 0; t < N_T; ++t) acp[t] = (t == 0) ? 1.0 : acum[t - 1];
  }
  double sqrt_recip[N_T], sqrt_recipm1[N_T], log_var[N_T], coef1[N_T], coef2[N_T];
  for (int t = 0; t < N_T; ++t) {
    sqrt_recip[t]   = sqrt(1.0 / acum[t]);
    sqrt_recipm1[t] = sqrt(1.0 / acum[t] - 1.0);
    double pv = betas[t] * (1.0 - acp[t]) / (1.0 - acum[t]);
    log_var[t] = log(pv > 1e-20 ? pv : 1e-20);
    coef1[t] = betas[t] * sqrt(acp[t]) / (1.0 - acum[t]);
    coef2[t] = (1.0 - acp[t]) * sqrt(alphas[t]) / (1.0 - acum[t]);
  }

  // denoising loop: i = 9..0
  for (int step = 0; step < N_T; ++step) {
    int i = N_T - 1 - step;
    const float* xaSrc = (step == 0) ? x_init : (const float*)xa;
    gemm256_k<AM_CAT, ACT_MISH><<<g, 256, 0, stream>>>(
        nullptr, nullptr, xaSrc, te_all + i * 16, hb16, 320, m1w16, m1b, a0);
    gemm256_k<AM_F16, ACT_MISH><<<g, 256, 0, stream>>>(
        nullptr, a0, nullptr, nullptr, nullptr, 256, m2w16, m2b, a1);
    gemm256_k<AM_F16, ACT_MISH><<<g, 256, 0, stream>>>(
        nullptr, a1, nullptr, nullptr, nullptr, 256, m3w16, m3b, a0);
    float sigma = (i > 0) ? (float)exp(0.5 * log_var[i]) : 0.f;
    fin_k<<<g, 256, 0, stream>>>(
        a0, finw16, finb, xaSrc, step_noise + (size_t)step * BATCH * N_ACT,
        xa, q_out,
        (float)sqrt_recip[i], (float)sqrt_recipm1[i],
        (float)coef1[i], (float)coef2[i], sigma, (i == 0) ? 1 : 0);
  }

  tail_k<<<(BATCH * N_ACT + 255) / 256, 256, 0, stream>>>(
      step_noise + (size_t)(N_T - 1) * BATCH * N_ACT, qlog, mask, noiseo,
      (float)log_var[0]);
}
